// SimplifiedLinearAttention_65180423685675
// MI455X (gfx1250) — matrix operand order repaired and checked by reading, not compiled, not run on hardware
//
#include <hip/hip_runtime.h>

// ---- types ----------------------------------------------------------------
typedef __attribute__((ext_vector_type(16))) __bf16 v16bf;
typedef __attribute__((ext_vector_type(8)))  __bf16 v8bf;
typedef __attribute__((ext_vector_type(4)))  __bf16 v4bf;
typedef __attribute__((ext_vector_type(8)))  float  v8f;

union U16 { v16bf v; v8bf h[2]; };

// Problem constants (fixed by the reference setup)
#define Bc   16
#define Nc   3136
#define Dc   512
#define NHc  8
#define HDc  64
#define BNc  (Bc * Nc)      // 50176
#define HHc  56
#define WWc  56

// A-fragment: 16x32 bf16, lane m = lane&15, K contiguous {h*16..h*16+15}
static __device__ __forceinline__ v16bf ldA(const __bf16* p) {
  U16 u;
  u.h[0] = *(const v8bf*)(p);
  u.h[1] = *(const v8bf*)(p + 8);
  return u.v;
}

static __device__ __forceinline__ v8f wmma_bf16(v16bf a, v16bf b, v8f c) {
  return __builtin_amdgcn_wmma_f32_16x16x32_bf16(false, a, false, b, (short)0, c,
                                                 false, false);
}

// ---- conversion kernels ---------------------------------------------------
__global__ void k_cvt_x(const float* __restrict__ x, __bf16* __restrict__ xb) {
  size_t i = ((size_t)blockIdx.x * 256 + threadIdx.x) * 4;
  float4 f = *(const float4*)(x + i);
  v4bf o;
  o[0] = (__bf16)f.x; o[1] = (__bf16)f.y; o[2] = (__bf16)f.z; o[3] = (__bf16)f.w;
  *(v4bf*)(xb + i) = o;
}

// Wqkvt[n, k] = combined W[k, n], n in [0,1536), k in [0,512)
__global__ void k_cvt_wqkv(const float* __restrict__ Wq, const float* __restrict__ Wkv,
                           __bf16* __restrict__ wt) {
  int idx = blockIdx.x * 256 + threadIdx.x;      // < 1536*512
  int n = idx >> 9, kk = idx & 511;
  float v = (n < Dc) ? Wq[(size_t)kk * Dc + n]
                     : Wkv[(size_t)kk * (2 * Dc) + (n - Dc)];
  wt[idx] = (__bf16)v;
}

__global__ void k_cvt_wp(const float* __restrict__ Wp, __bf16* __restrict__ wt) {
  int idx = blockIdx.x * 256 + threadIdx.x;      // < 512*512
  int n = idx >> 9, kk = idx & 511;
  wt[idx] = (__bf16)Wp[(size_t)kk * Dc + n];
}

// ---- GEMM1: X(50176x512) * W(512x1536) ------------------------------------
// q -> row-major (feeds WMMA A-frags + d_inv);
// k,v -> per-head transposed kT/vT[(b*8+h)*64+d][n] via vectorized v8bf stores
// (WMMA C layout has M contiguous in a lane's VGPRs -> transposed store is 16B).
__global__ void k_gemm_qkv(const __bf16* __restrict__ xb, const __bf16* __restrict__ wt,
                           const float* __restrict__ pos,
                           __bf16* __restrict__ qb, __bf16* __restrict__ kT,
                           __bf16* __restrict__ vT) {
  int wg   = blockIdx.x * 8 + (threadIdx.x >> 5);   // 0..75263
  int lane = threadIdx.x & 31, l15 = lane & 15, hs = lane >> 4;
  int mtile = wg / 24, strip = wg % 24;
  int mbase = mtile * 16, cbase = strip * 64;

  const __bf16* arow = xb + (size_t)(mbase + l15) * Dc;
  v8f acc[4];
#pragma unroll
  for (int t = 0; t < 4; ++t)
#pragma unroll
    for (int i = 0; i < 8; ++i) acc[t][i] = 0.f;

  for (int kk = 0; kk < Dc; kk += 32) {
    __builtin_prefetch(arow + kk + 64, 0, 3);       // global_prefetch_b8 (near)
    v16bf a = ldA(arow + kk + hs * 16);
#pragma unroll
    for (int t = 0; t < 4; ++t) {
      const __bf16* bptr = wt + (size_t)(cbase + t * 16 + l15) * Dc + kk + hs * 16;
      v16bf b = *(const v16bf*)bptr;
      acc[t] = wmma_bf16(a, b, acc[t]);
    }
  }

  int b0 = mbase / Nc;              // 16-row tile never crosses a batch (3136%16==0)
  int n0 = (mbase % Nc) + hs * 8;   // n contiguous over r
#pragma unroll
  for (int t = 0; t < 4; ++t) {
    int nout = cbase + t * 16 + l15;
    if (nout < Dc) {                               // ---- q: row-major, ReLU
#pragma unroll
      for (int r = 0; r < 8; ++r) {
        int row = mbase + hs * 8 + r;
        qb[(size_t)row * Dc + nout] = (__bf16)fmaxf(acc[t][r], 0.f);
      }
    } else if (nout < 2 * Dc) {                    // ---- k: +pos, ReLU, transposed
      int d2 = nout - Dc;
      int hh = d2 >> 6, ee = d2 & 63;
      v8bf o;
#pragma unroll
      for (int r = 0; r < 8; ++r) {
        float val = acc[t][r] + pos[(size_t)(n0 + r) * Dc + d2];
        o[r] = (__bf16)fmaxf(val, 0.f);
      }
      *(v8bf*)(kT + ((size_t)(b0 * NHc + hh) * HDc + ee) * Nc + n0) = o;
    } else {                                       // ---- v: transposed
      int d2 = nout - 2 * Dc;
      int hh = d2 >> 6, ee = d2 & 63;
      v8bf o;
#pragma unroll
      for (int r = 0; r < 8; ++r) o[r] = (__bf16)acc[t][r];
      *(v8bf*)(vT + ((size_t)(b0 * NHc + hh) * HDc + ee) * Nc + n0) = o;
    }
  }
}

// ---- k_sum over contiguous kT rows ----------------------------------------
__global__ void k_ksum(const __bf16* __restrict__ kT, float* __restrict__ ks) {
  __shared__ float sd[256];
  int bh = blockIdx.x;
  int t = threadIdx.x, d = t >> 2, seg = t & 3;
  const __bf16* row = kT + ((size_t)bh * HDc + d) * Nc;
  float s = 0.f;
  for (int n = seg; n < Nc; n += 4) s += (float)row[n];
  sd[t] = s;
  __syncthreads();
  if (t < 64) ks[bh * 64 + t] = sd[4 * t] + sd[4 * t + 1] + sd[4 * t + 2] + sd[4 * t + 3];
}

// ---- d_inv ----------------------------------------------------------------
__global__ void k_dinv(const __bf16* __restrict__ qb, const float* __restrict__ ks,
                       float* __restrict__ dinv) {
  int idx = blockIdx.x * 256 + threadIdx.x;       // < 128*3136
  int bh = idx / Nc, n = idx % Nc;
  int b = bh >> 3, h = bh & 7;
  const __bf16* qr = qb + (size_t)(b * Nc + n) * Dc + h * HDc;
  const float* kr = ks + bh * 64;
  float s = 0.f;
#pragma unroll 8
  for (int d = 0; d < HDc; ++d) s += (float)qr[d] * kr[d];
  dinv[idx] = 1.f / (s + 1e-6f);
}

// ---- ctx = K^T V per (b,h): both operands now contiguous b128 loads -------
__global__ void k_ctx(const __bf16* __restrict__ kT, const __bf16* __restrict__ vT,
                      __bf16* __restrict__ ctxT) {
  int bh = blockIdx.x;
  int w = threadIdx.x >> 5, lane = threadIdx.x & 31, l15 = lane & 15, hs = lane >> 4;
  int dbase = w * 16;

  const __bf16* arow = kT + ((size_t)bh * HDc + dbase + l15) * Nc;  // A[m=d][k=n]
  v8f acc[4];
#pragma unroll
  for (int t = 0; t < 4; ++t)
#pragma unroll
    for (int i = 0; i < 8; ++i) acc[t][i] = 0.f;

  for (int kk = 0; kk < Nc; kk += 32) {
    __builtin_prefetch(arow + kk + 64, 0, 3);
    v16bf a = ldA(arow + kk + hs * 16);
#pragma unroll
    for (int t = 0; t < 4; ++t) {
      const __bf16* bptr = vT + ((size_t)bh * HDc + t * 16 + l15) * Nc + kk + hs * 16;
      v16bf b = *(const v16bf*)bptr;                   // B[k=n][col=e] = vT[e][n]
      acc[t] = wmma_bf16(a, b, acc[t]);
    }
  }

#pragma unroll
  for (int t = 0; t < 4; ++t) {                        // ctxT[e][d], d contiguous
    int e = t * 16 + l15;
    v8bf o;
#pragma unroll
    for (int r = 0; r < 8; ++r) o[r] = (__bf16)acc[t][r];
    *(v8bf*)(ctxT + ((size_t)bh * HDc + e) * HDc + dbase + hs * 8) = o;
  }
}

// ---- attn = (Q*ctx)*d_inv + depthwise5x5(vT), written bf16 ----------------
__global__ void k_attn(const __bf16* __restrict__ qb, const __bf16* __restrict__ ctxT,
                       const float* __restrict__ dinv, const __bf16* __restrict__ vT,
                       const float* __restrict__ dw, const float* __restrict__ db,
                       __bf16* __restrict__ yb) {
  int wg   = blockIdx.x * 8 + (threadIdx.x >> 5);  // 0..25087
  int lane = threadIdx.x & 31, l15 = lane & 15, hs = lane >> 4;
  int bh = wg / 196, mt = wg % 196;
  int b = bh >> 3, h = bh & 7;
  int mbase = mt * 16;

  const __bf16* arow = qb + (size_t)(b * Nc + mbase + l15) * Dc + h * HDc;
  v8f acc[4];
#pragma unroll
  for (int t = 0; t < 4; ++t)
#pragma unroll
    for (int i = 0; i < 8; ++i) acc[t][i] = 0.f;

#pragma unroll
  for (int kk = 0; kk < HDc; kk += 32) {
    v16bf a = ldA(arow + kk + hs * 16);
#pragma unroll
    for (int t = 0; t < 4; ++t) {
      const __bf16* bptr = ctxT + ((size_t)bh * HDc + t * 16 + l15) * HDc + kk + hs * 16;
      v16bf bf = *(const v16bf*)bptr;
      acc[t] = wmma_bf16(a, bf, acc[t]);
    }
  }

#pragma unroll
  for (int t = 0; t < 4; ++t) {
    int e = t * 16 + l15;
    const float* wch = dw + e * 25;
    float bias = db[e];
    const __bf16* vrow = vT + ((size_t)bh * HDc + e) * Nc;   // L2-resident
#pragma unroll
    for (int r = 0; r < 8; ++r) {
      int n = mbase + hs * 8 + r;
      float di = dinv[(size_t)bh * Nc + n];
      int ys = n / WWc, xs = n % WWc;
      float conv = bias;
      for (int dy = 0; dy < 5; ++dy) {
        int yy = ys + dy - 2;
        if ((unsigned)yy >= HHc) continue;
        const __bf16* vr2 = vrow + yy * WWc;
        for (int dx = 0; dx < 5; ++dx) {
          int xx = xs + dx - 2;
          if ((unsigned)xx >= WWc) continue;
          conv += wch[dy * 5 + dx] * (float)vr2[xx];
        }
      }
      float val = acc[t][r] * di + conv;
      yb[(size_t)(b * Nc + n) * Dc + h * HDc + e] = (__bf16)val;
    }
  }
}

// ---- final GEMM: y(50176x512) * Wp(512x512) + bp -> f32 out ---------------
__global__ void k_gemm_out(const __bf16* __restrict__ yb, const __bf16* __restrict__ wt,
                           const float* __restrict__ bias, float* __restrict__ out) {
  int wg   = blockIdx.x * 8 + (threadIdx.x >> 5);  // 0..25087
  int lane = threadIdx.x & 31, l15 = lane & 15, hs = lane >> 4;
  int mtile = wg >> 3, cbase = (wg & 7) * 64;
  int mbase = mtile * 16;

  const __bf16* arow = yb + (size_t)(mbase + l15) * Dc;
  v8f acc[4];
#pragma unroll
  for (int t = 0; t < 4; ++t)
#pragma unroll
    for (int i = 0; i < 8; ++i) acc[t][i] = 0.f;

  for (int kk = 0; kk < Dc; kk += 32) {
    __builtin_prefetch(arow + kk + 64, 0, 3);
    v16bf a = ldA(arow + kk + hs * 16);
#pragma unroll
    for (int t = 0; t < 4; ++t) {
      const __bf16* bptr = wt + (size_t)(cbase + t * 16 + l15) * Dc + kk + hs * 16;
      v16bf b = *(const v16bf*)bptr;
      acc[t] = wmma_bf16(a, b, acc[t]);
    }
  }

#pragma unroll
  for (int t = 0; t < 4; ++t) {
    int col = cbase + t * 16 + l15;
    float bb = bias[col];
#pragma unroll
    for (int r = 0; r < 8; ++r) {
      int row = mbase + hs * 8 + r;
      out[(size_t)row * Dc + col] = acc[t][r] + bb;
    }
  }
}

// ---- host launcher --------------------------------------------------------
extern "C" void kernel_launch(void* const* d_in, const int* in_sizes, int n_in,
                              void* d_out, int out_size, void* d_ws, size_t ws_size,
                              hipStream_t stream) {
  (void)in_sizes; (void)n_in; (void)out_size; (void)ws_size;
  const float* x    = (const float*)d_in[0];
  const float* Wq   = (const float*)d_in[1];
  const float* Wkv  = (const float*)d_in[2];
  const float* pos  = (const float*)d_in[3];
  const float* dw   = (const float*)d_in[4];
  const float* db   = (const float*)d_in[5];
  const float* Wp   = (const float*)d_in[6];
  const float* bp   = (const float*)d_in[7];
  float* out = (float*)d_out;

  const size_t eBN = (size_t)BNc * Dc;            // 25,690,112 elements
  char* ws = (char*)d_ws;
  size_t off = 0;
  auto carve = [&](size_t bytes) {
    off = (off + 255) & ~(size_t)255;
    void* p = ws + off;
    off += bytes;
    return p;
  };
  __bf16* xb    = (__bf16*)carve(eBN * 2);
  __bf16* qb    = (__bf16*)carve(eBN * 2);
  __bf16* kT    = (__bf16*)carve(eBN * 2);
  __bf16* vT    = (__bf16*)carve(eBN * 2);
  __bf16* yb    = (__bf16*)carve(eBN * 2);
  __bf16* wqkvt = (__bf16*)carve((size_t)1536 * 512 * 2);
  __bf16* wpt   = (__bf16*)carve((size_t)512 * 512 * 2);
  __bf16* ctxT  = (__bf16*)carve((size_t)Bc * NHc * HDc * HDc * 2);
  float*  ks    = (float*)carve((size_t)Bc * NHc * HDc * 4);
  float*  dinv  = (float*)carve((size_t)Bc * NHc * Nc * 4);

  k_cvt_x    <<<25088, 256, 0, stream>>>(x, xb);
  k_cvt_wqkv <<<3072,  256, 0, stream>>>(Wq, Wkv, wqkvt);
  k_cvt_wp   <<<1024,  256, 0, stream>>>(Wp, wpt);
  k_gemm_qkv <<<9408,  256, 0, stream>>>(xb, wqkvt, pos, qb, kT, vT);
  k_ksum     <<<Bc * NHc, 256, 0, stream>>>(kT, ks);
  k_dinv     <<<1568,  256, 0, stream>>>(qb, ks, dinv);
  k_ctx      <<<Bc * NHc, 128, 0, stream>>>(kT, vT, ctxT);
  k_attn     <<<3136,  256, 0, stream>>>(qb, ctxT, dinv, vT, dw, db, yb);
  k_gemm_out <<<3136,  256, 0, stream>>>(yb, wpt, bp, out);
}
